// block_seq_75960791597885
// MI455X (gfx1250) — compile-verified, matrix operand order checked by reading, not run on hardware
//
#include <hip/hip_runtime.h>

// ---------------------------------------------------------------------------
// MI455X (gfx1250) implementation.
// The three 3x3 convs (~60 GFLOP) are compute bound (min HBM traffic ~134MB
// @ 23.3TB/s ~ 6us), so they run as implicit GEMM on
// v_wmma_f32_16x16x32_bf16 (fp32 accumulate). Intermediates (~300MB) are
// largely L2-resident (192MB), so elementwise/stats stages are cheap.
// LDS staging uses GLOBAL_LOAD_ASYNC_TO_LDS_B128 (ASYNCcnt-tracked, no VGPR
// round trip) -- builtin signature probed: (v4i addrspace(1)*, v4i
// addrspace(3)*, imm offset, imm cpol).
// ---------------------------------------------------------------------------

typedef __attribute__((ext_vector_type(16))) __bf16 v16bf;
typedef __attribute__((ext_vector_type(8)))  float  v8f;
typedef int vi4 __attribute__((vector_size(16)));

struct __align__(16) U4 { unsigned int x, y, z, w; };
struct __align__(16) F4 { float x, y, z, w; };

#define NN   4
#define CC   64
#define HH   256
#define WW   256
#define HWP  65536          // H*W
#define CF   72             // G*K*K
#define CFP  80             // padded to 5 WMMA n-tiles
#define KS   576            // 9*64 GEMM K for 3x3 convs
#define EPSV 1e-5f

#if defined(__has_builtin)
#if __has_builtin(__builtin_amdgcn_global_load_async_to_lds_b128)
#define HAVE_ASYNC_LDS 1
#endif
#endif
#ifndef HAVE_ASYNC_LDS
#define HAVE_ASYNC_LDS 0
#endif

#if HAVE_ASYNC_LDS
typedef __attribute__((address_space(1))) vi4* gvec_p;
typedef __attribute__((address_space(3))) vi4* lvec_p;
#endif

static __device__ __forceinline__ unsigned short f32_to_bf16(float f) {
    unsigned int u = __float_as_uint(f);
    unsigned int r = u + 0x7FFFu + ((u >> 16) & 1u);   // round-nearest-even
    return (unsigned short)(r >> 16);
}
static __device__ __forceinline__ float lk(float v) { return v >= 0.f ? v : 0.01f * v; }

static __device__ __forceinline__ v16bf load16(const unsigned short* p0,
                                               const unsigned short* p1) {
    union { U4 q[2]; v16bf v; } u;
    u.q[0] = *(const U4*)p0;
    u.q[1] = *(const U4*)p1;
    return u.v;
}

// ------------------------- weight packing ----------------------------------
// dst[co][k], k = tap*64 + ci, rows padded to coutpad (zeros).
__global__ __launch_bounds__(256) void pack3x3(const float* __restrict__ src,
                                               unsigned short* __restrict__ dst,
                                               int cout, int coutpad) {
    int idx = blockIdx.x * 256 + threadIdx.x;
    if (idx >= coutpad * KS) return;
    int co = idx / KS, k = idx - co * KS;
    int t = k >> 6, ci = k & 63;
    int ky = t / 3, kx = t - ky * 3;
    float v = (co < cout) ? src[(((size_t)co * CC + ci) * 3 + ky) * 3 + kx] : 0.f;
    dst[idx] = f32_to_bf16(v);
}

__global__ __launch_bounds__(256) void pack1x1(const float* __restrict__ src,
                                               unsigned short* __restrict__ dst) {
    int idx = blockIdx.x * 256 + threadIdx.x;
    if (idx >= CC * CC) return;
    dst[idx] = f32_to_bf16(src[idx]);
}

// ------------------------- NCHW f32 -> NHWC bf16 ---------------------------
__global__ __launch_bounds__(256) void cvt_nhwc(const float* __restrict__ src,
                                                unsigned short* __restrict__ dst) {
    int gid = blockIdx.x * 256 + threadIdx.x;   // pixel id over N*H*W
    int n = gid >> 16, pix = gid & 65535;
    U4 b4[8];
    unsigned short* buf = (unsigned short*)b4;
#pragma unroll
    for (int c = 0; c < CC; c++)
        buf[c] = f32_to_bf16(src[(((size_t)n * CC + c) << 16) + pix]);
    U4* dp = (U4*)(dst + ((size_t)gid << 6));
#pragma unroll
    for (int i = 0; i < 8; i++) dp[i] = b4[i];
}

// ------------------------- zero small stats --------------------------------
__global__ void zero_stats(float* stats, float* stats2) {
    int t = threadIdx.x;
    if (t < 2 * CF) stats[t] = 0.f;
    if (t < 2 * 128) stats2[t] = 0.f;
}

// ------------------------- 1x1 conv (residual) -----------------------------
// out[n,co,pix] = sum_ci x[n,pix,ci]*wr[co,ci] + br[co]
__global__ __launch_bounds__(256) void conv1x1_res(const unsigned short* __restrict__ xb,
                                                   const unsigned short* __restrict__ wrb,
                                                   const float* __restrict__ br,
                                                   float* __restrict__ out) {
    const int n = blockIdx.z;
    const int wave = threadIdx.x >> 5, lane = threadIdx.x & 31;
    const int pix0 = blockIdx.x * 128 + wave * 16;
    const int m = lane & 15;
    const int khalf = (lane & 16) ? 8 : 0;

    v8f acc[4];
#pragma unroll
    for (int j = 0; j < 4; j++)
#pragma unroll
        for (int i = 0; i < 8; i++) acc[j][i] = 0.f;

#pragma unroll
    for (int kc = 0; kc < 64; kc += 32) {
        const unsigned short* ap = xb + (((size_t)n << 16) + pix0 + m) * CC + kc + khalf;
        v16bf a = load16(ap, ap + 16);
        v16bf bf[4];
#pragma unroll
        for (int j = 0; j < 4; j++) {
            const unsigned short* bp = wrb + (size_t)(j * 16 + m) * CC + kc + khalf;
            bf[j] = load16(bp, bp + 16);
        }
#pragma unroll
        for (int j = 0; j < 4; j++)
            acc[j] = __builtin_amdgcn_wmma_f32_16x16x32_bf16(false, a, false, bf[j],
                                                             (short)0, acc[j], false, false);
    }
    const int col = lane & 15;
    const int m0 = (lane >> 4) * 8;
#pragma unroll
    for (int j = 0; j < 4; j++) {
        int co = j * 16 + col;
        float bv = br[co];
        float* op = out + (((size_t)n * CC + co) << 16) + pix0 + m0;
        v8f c = acc[j];
        F4 lo{c[0] + bv, c[1] + bv, c[2] + bv, c[3] + bv};
        F4 hi{c[4] + bv, c[5] + bv, c[6] + bv, c[7] + bv};
        *(F4*)op = lo;
        *(F4*)(op + 4) = hi;
    }
}

// ------------------------- 3x3 conv via WMMA -------------------------------
// NT: number of 16-wide Cout tiles (4 -> Cout 64, 5 -> Cout 72 padded to 80)
// PADMODE: 0 = zero pad, 1 = reflect pad
// MODE: 0 = +bias, write f32 NCHW and bf16 NHWC copy        (conv1)
//       1 = no bias, write f32 NCHW (Cout=72)               (wf conv)
//       2 = out = res + leaky(acc + bias), RMW on out       (conv2 + residual)
template <int NT, int PADMODE, int MODE>
__global__ __launch_bounds__(256) void conv3x3_wmma(const unsigned short* __restrict__ inb,
                                                    const unsigned short* __restrict__ wb,
                                                    const float* __restrict__ bias,
                                                    float* __restrict__ out,
                                                    unsigned short* __restrict__ outb) {
    constexpr int COUT = (MODE == 1) ? CF : CC;
    __shared__ __attribute__((aligned(16))) unsigned short smem[3][132][72];

    const int tid = threadIdx.x;
    const int n = blockIdx.z, y = blockIdx.y;
    const int x0 = blockIdx.x * 128;

    // warm weights into cache while staging activations
    for (int i = tid; i < NT * 16; i += 256)
        __builtin_prefetch(wb + (size_t)i * KS, 0, 3);

    // stage 3 rows x 130 px x 64ch bf16 into LDS (pixel stride padded to 72)
    for (int idx = tid; idx < 3 * 130 * 8; idx += 256) {
        int r = idx / (130 * 8);
        int rem = idx - r * 130 * 8;
        int px = rem >> 3, cq = rem & 7;
        int ys = y + r - 1, xs = x0 - 1 + px;
        bool ok;
        if (PADMODE == 1) {  // reflect
            if (ys < 0) ys = -ys;
            if (ys > HH - 1) ys = 2 * (HH - 1) - ys;
            if (xs < 0) xs = -xs;
            if (xs > WW - 1) xs = 2 * (WW - 1) - xs;
            ok = true;
        } else {
            ok = (ys >= 0 && ys < HH && xs >= 0 && xs < WW);
        }
        unsigned short* lp = &smem[r][px][cq * 8];
        if (ok) {
            const unsigned short* gp =
                inb + ((((size_t)n * HH + ys) * WW + xs) << 6) + cq * 8;
#if HAVE_ASYNC_LDS
            __builtin_amdgcn_global_load_async_to_lds_b128((gvec_p)gp, (lvec_p)lp, 0, 0);
#else
            *(U4*)lp = *(const U4*)gp;
#endif
        } else {
            *(U4*)lp = U4{0, 0, 0, 0};
        }
    }
#if HAVE_ASYNC_LDS
    asm volatile("s_wait_asynccnt 0x0" ::: "memory");
#endif
    __syncthreads();

    const int wave = tid >> 5, lane = tid & 31;
    const int waveM = wave * 16;
    const int m = lane & 15;               // A row (pixel) and B col (cout) idx
    const int khalf = (lane & 16) ? 8 : 0; // lane-half K offset per 16-bit layout

    v8f acc[NT];
#pragma unroll
    for (int j = 0; j < NT; j++)
#pragma unroll
        for (int i = 0; i < 8; i++) acc[j][i] = 0.f;

#pragma unroll
    for (int t = 0; t < 9; t++) {
        const int ty = t / 3, tx = t - ty * 3;
        const unsigned short* arow = &smem[ty][waveM + m + tx][0];
#pragma unroll
        for (int kc = 0; kc < 64; kc += 32) {
            v16bf a = load16(arow + kc + khalf, arow + kc + khalf + 16);
            // batch all B fragments first -> one load clause, then WMMA chain
            v16bf bf[NT];
#pragma unroll
            for (int j = 0; j < NT; j++) {
                const unsigned short* wrow =
                    wb + (size_t)(j * 16 + m) * KS + t * 64 + kc + khalf;
                bf[j] = load16(wrow, wrow + 16);
            }
#pragma unroll
            for (int j = 0; j < NT; j++)
                acc[j] = __builtin_amdgcn_wmma_f32_16x16x32_bf16(
                    false, a, false, bf[j], (short)0, acc[j], false, false);
        }
    }

    // epilogue: C/D layout — lane holds col = lane&15, pixels m0..m0+7
    const int col = lane & 15;
    const int m0 = (lane >> 4) * 8;
    const size_t pixrow = (size_t)y * WW + x0 + waveM + m0;  // offset within image
#pragma unroll
    for (int j = 0; j < NT; j++) {
        const int co = j * 16 + col;
        v8f c = acc[j];
        if (MODE == 1) {
            if (co < CF) {
                float* op = out + (((size_t)n * COUT + co) << 16) + pixrow;
                *(F4*)op = F4{c[0], c[1], c[2], c[3]};
                *(F4*)(op + 4) = F4{c[4], c[5], c[6], c[7]};
            }
        } else if (MODE == 0) {
            const float bv = bias[co];
            float* op = out + (((size_t)n * COUT + co) << 16) + pixrow;
            *(F4*)op = F4{c[0] + bv, c[1] + bv, c[2] + bv, c[3] + bv};
            *(F4*)(op + 4) = F4{c[4] + bv, c[5] + bv, c[6] + bv, c[7] + bv};
#pragma unroll
            for (int r = 0; r < 8; r++) {
                size_t pp = ((size_t)n << 16) + pixrow + r;
                outb[(pp << 6) + co] = f32_to_bf16(c[r] + bv);
            }
        } else {  // MODE 2: final = res + leaky(acc + bias)
            const float bv = bias[co];
            float* op = out + (((size_t)n * COUT + co) << 16) + pixrow;
            F4 r0 = *(const F4*)op;
            F4 r1 = *(const F4*)(op + 4);
            *(F4*)op = F4{r0.x + lk(c[0] + bv), r0.y + lk(c[1] + bv),
                          r0.z + lk(c[2] + bv), r0.w + lk(c[3] + bv)};
            *(F4*)(op + 4) = F4{r1.x + lk(c[4] + bv), r1.y + lk(c[5] + bv),
                                r1.z + lk(c[6] + bv), r1.w + lk(c[7] + bv)};
        }
    }
}

// ------------------------- BN stats over (N,H,W) per channel ---------------
__global__ __launch_bounds__(256) void bn_stats(const float* __restrict__ sig,
                                                float* __restrict__ stats) {
    __shared__ float red[256];
    const int c = blockIdx.y;
    const int tid = threadIdx.x;
    const size_t e0 = (size_t)blockIdx.x * 4096;
    float s = 0.f, q = 0.f;
    for (int i = tid; i < 4096; i += 256) {
        size_t e = e0 + i;
        size_t nn = e >> 16, pix = e & 65535;
        float v = sig[((nn * CF + c) << 16) + pix];
        s += v;
        q += v * v;
    }
    red[tid] = s;
    __syncthreads();
    for (int o = 128; o > 0; o >>= 1) {
        if (tid < o) red[tid] += red[tid + o];
        __syncthreads();
    }
    if (tid == 0) atomicAdd(&stats[2 * c], red[0]);
    __syncthreads();
    red[tid] = q;
    __syncthreads();
    for (int o = 128; o > 0; o >>= 1) {
        if (tid < o) red[tid] += red[tid + o];
        __syncthreads();
    }
    if (tid == 0) atomicAdd(&stats[2 * c + 1], red[0]);
}

// ------------------------- freq filter: norm + softmax + 9-tap -------------
__global__ __launch_bounds__(256) void freqfilter(const float* __restrict__ sig,
                                                  const float* __restrict__ stats,
                                                  const float* __restrict__ gamma,
                                                  const float* __restrict__ beta,
                                                  const float* __restrict__ y1,
                                                  float* __restrict__ y2) {
    const int gid = blockIdx.x * 256 + threadIdx.x;  // pixel over N*H*W
    const int n = gid >> 16, pix = gid & 65535;
    const int h = pix >> 8, w = pix & 255;
    const float icnt = 1.0f / 262144.0f;  // N*H*W

    float s[CF];
    float mx = -1e30f;
#pragma unroll
    for (int c = 0; c < CF; c++) {
        float su = stats[2 * c], sq = stats[2 * c + 1];
        float mu = su * icnt;
        float var = sq * icnt - mu * mu;
        float v = sig[(((size_t)n * CF + c) << 16) + pix];
        v = (v - mu) * rsqrtf(var + EPSV) * gamma[c] + beta[c];
        s[c] = v;
        mx = fmaxf(mx, v);
    }
    float den = 0.f;
#pragma unroll
    for (int c = 0; c < CF; c++) {
        float e = __expf(s[c] - mx);
        s[c] = e;
        den += e;
    }
    const float inv = 1.0f / den;

    int hm = h - 1; if (hm < 0) hm = 1;
    int hp = h + 1; if (hp > HH - 1) hp = HH - 2;
    int wm = w - 1; if (wm < 0) wm = 1;
    int wp = w + 1; if (wp > WW - 1) wp = WW - 2;
    const int hh[3] = {hm, h, hp};
    const int ww[3] = {wm, w, wp};

#pragma unroll
    for (int g = 0; g < 8; g++) {
        float kk[9];
#pragma unroll
        for (int t = 0; t < 9; t++) kk[t] = s[g * 9 + t] * inv;
#pragma unroll
        for (int cg = 0; cg < 8; cg++) {
            const int ch = g * 8 + cg;
            const float* b = y1 + (((size_t)n * CC + ch) << 16);
            float acc = b[h * WW + w];  // delta at center tap
#pragma unroll
            for (int t = 0; t < 9; t++)
                acc -= b[hh[t / 3] * WW + ww[t % 3]] * kk[t];
            y2[(((size_t)n * CC + ch) << 16) + pix] = acc;
        }
    }
}

// ------------------------- half-instance-norm stats ------------------------
__global__ __launch_bounds__(256) void hin_stats(const float* __restrict__ y2,
                                                 float* __restrict__ stats2) {
    __shared__ float red[256];
    const int idx = blockIdx.x;  // n*32 + c, c in [0,32)
    const int tid = threadIdx.x;
    const float* b = y2 + (((size_t)(idx >> 5) * CC + (idx & 31)) << 16);
    float s = 0.f, q = 0.f;
    for (int i = tid; i < HWP; i += 256) {
        float v = b[i];
        s += v;
        q += v * v;
    }
    red[tid] = s;
    __syncthreads();
    for (int o = 128; o > 0; o >>= 1) {
        if (tid < o) red[tid] += red[tid + o];
        __syncthreads();
    }
    if (tid == 0) stats2[2 * idx] = red[0];
    __syncthreads();
    red[tid] = q;
    __syncthreads();
    for (int o = 128; o > 0; o >>= 1) {
        if (tid < o) red[tid] += red[tid + o];
        __syncthreads();
    }
    if (tid == 0) stats2[2 * idx + 1] = red[0];
}

// ------------------------- hin apply + leaky -> NHWC bf16 ------------------
__global__ __launch_bounds__(256) void hin_apply(const float* __restrict__ y2,
                                                 const float* __restrict__ stats2,
                                                 unsigned short* __restrict__ y3b) {
    const int gid = blockIdx.x * 256 + threadIdx.x;  // pixel over N*H*W
    const int n = gid >> 16, pix = gid & 65535;
    const float ic = 1.0f / 65536.0f;
    U4 b4[8];
    unsigned short* buf = (unsigned short*)b4;
#pragma unroll
    for (int c = 0; c < CC; c++) {
        float v = y2[(((size_t)n * CC + c) << 16) + pix];
        if (c < 32) {
            int si = n * 32 + c;
            float mu = stats2[2 * si] * ic;
            float var = stats2[2 * si + 1] * ic - mu * mu;
            v = (v - mu) * rsqrtf(var + EPSV);
        }
        buf[c] = f32_to_bf16(lk(v));
    }
    U4* dp = (U4*)(y3b + ((size_t)gid << 6));
#pragma unroll
    for (int i = 0; i < 8; i++) dp[i] = b4[i];
}

// ---------------------------------------------------------------------------
extern "C" void kernel_launch(void* const* d_in, const int* in_sizes, int n_in,
                              void* d_out, int out_size, void* d_ws, size_t ws_size,
                              hipStream_t stream) {
    const float* x     = (const float*)d_in[0];
    const float* w1    = (const float*)d_in[1];
    const float* b1    = (const float*)d_in[2];
    const float* wf    = (const float*)d_in[3];
    const float* gamma = (const float*)d_in[4];
    const float* beta  = (const float*)d_in[5];
    const float* w2    = (const float*)d_in[6];
    const float* b2    = (const float*)d_in[7];
    const float* wr    = (const float*)d_in[8];
    const float* br    = (const float*)d_in[9];
    float* out = (float*)d_out;

    char* ws = (char*)d_ws;
    // workspace layout (~296 MiB, all 16B aligned)
    unsigned short* XB   = (unsigned short*)(ws + 0ULL);          // x NHWC bf16
    unsigned short* Y1B  = (unsigned short*)(ws + 33554432ULL);   // y1 NHWC bf16
    unsigned short* Y3B  = (unsigned short*)(ws + 67108864ULL);   // y3 NHWC bf16
    float* Y1            = (float*)(ws + 100663296ULL);           // conv1 out f32
    float* Y2            = (float*)(ws + 167772160ULL);           // freqfilter out
    float* SIG           = (float*)(ws + 234881024ULL);           // wf conv out (72ch)
    unsigned short* W1B  = (unsigned short*)(ws + 310378496ULL);
    unsigned short* WFB  = (unsigned short*)(ws + 310452224ULL);
    unsigned short* W2B  = (unsigned short*)(ws + 310544384ULL);
    unsigned short* WRB  = (unsigned short*)(ws + 310618112ULL);
    float* STATS         = (float*)(ws + 310626304ULL);           // 72 x {sum,sumsq}
    float* STATS2        = (float*)(ws + 310626880ULL);           // 128 x {sum,sumsq}

    const dim3 convGrid(WW / 128, HH, NN);  // (2,256,4)

    zero_stats<<<1, 256, 0, stream>>>(STATS, STATS2);
    pack3x3<<<(CC * KS + 255) / 256, 256, 0, stream>>>(w1, W1B, CC, CC);
    pack3x3<<<(CFP * KS + 255) / 256, 256, 0, stream>>>(wf, WFB, CF, CFP);
    pack3x3<<<(CC * KS + 255) / 256, 256, 0, stream>>>(w2, W2B, CC, CC);
    pack1x1<<<(CC * CC + 255) / 256, 256, 0, stream>>>(wr, WRB);

    cvt_nhwc<<<NN * HWP / 256, 256, 0, stream>>>(x, XB);

    // residual path into d_out
    conv1x1_res<<<dim3(HWP / 128, 1, NN), 256, 0, stream>>>(XB, WRB, br, out);

    // y1 = conv1(x)+b1 (zero pad), also emit bf16 NHWC copy
    conv3x3_wmma<4, 0, 0><<<convGrid, 256, 0, stream>>>(XB, W1B, b1, Y1, Y1B);

    // sig = conv(wf) on reflect-padded y1
    conv3x3_wmma<5, 1, 1><<<convGrid, 256, 0, stream>>>(Y1B, WFB, nullptr, SIG, nullptr);

    bn_stats<<<dim3(64, CF), 256, 0, stream>>>(SIG, STATS);
    freqfilter<<<NN * HWP / 256, 256, 0, stream>>>(SIG, STATS, gamma, beta, Y1, Y2);

    hin_stats<<<NN * 32, 256, 0, stream>>>(Y2, STATS2);
    hin_apply<<<NN * HWP / 256, 256, 0, stream>>>(Y2, STATS2, Y3B);

    // out = res + leaky(conv2(y3)+b2)
    conv3x3_wmma<4, 0, 2><<<convGrid, 256, 0, stream>>>(Y3B, W2B, b2, out, nullptr);
}
